// ModulatedConv2DLayer_27754078667656
// MI455X (gfx1250) — compile-verified, matrix operand order checked by reading, not run on hardware
//
#include <hip/hip_runtime.h>

#define NB   8
#define CIN  512
#define FM   512
#define HH   64
#define WW   64
#define HW   4096
#define KT   9    // 3x3 taps
#define HP   66   // padded H
#define WP   66   // padded W

// StyleGAN2 wscale constants
#define COEF_S 0.04419417382415922f    // 1/sqrt(512)
#define COEF_W 0.014731391274719742f   // 1/sqrt(9*512)

typedef __attribute__((ext_vector_type(16))) __bf16        v16bf;
typedef __attribute__((ext_vector_type(8)))  float         v8f;
typedef __attribute__((ext_vector_type(8)))  unsigned int  v8u;

__device__ __forceinline__ unsigned short f2bf(float f) {
    unsigned u = __float_as_uint(f);
    u += 0x7FFFu + ((u >> 16) & 1u);          // round-to-nearest-even
    return (unsigned short)(u >> 16);
}

// Load 16 contiguous bf16 (32B) as one WMMA half-K fragment for this lane.
__device__ __forceinline__ v16bf load_frag16(const unsigned short* p) {
    const uint4* q = (const uint4*)p;
    uint4 lo = q[0];
    uint4 hi = q[1];
    v8u u;
    u[0] = lo.x; u[1] = lo.y; u[2] = lo.z; u[3] = lo.w;
    u[4] = hi.x; u[5] = hi.y; u[6] = hi.z; u[7] = hi.w;
    return __builtin_bit_cast(v16bf, u);
}

// ---------------- style: s[n][c] = (dlat @ mw)*COEF_S + bias + 1 ----------------
__global__ __launch_bounds__(256) void style_kernel(
    const float* __restrict__ dlat, const float* __restrict__ mw,
    const float* __restrict__ mb, float* __restrict__ s)
{
    int idx = blockIdx.x * 256 + threadIdx.x;     // 4096 total
    int n = idx >> 9, c = idx & 511;
    float acc = 0.f;
    for (int l = 0; l < 512; ++l)
        acc += dlat[n * 512 + l] * mw[l * 512 + c];
    s[idx] = acc * COEF_S + mb[c] + 1.0f;
}

__global__ __launch_bounds__(256) void zero_f(float* __restrict__ p) {
    p[blockIdx.x * 256 + threadIdx.x] = 0.f;
}

__global__ __launch_bounds__(256) void zero_u4(uint4* __restrict__ p) {
    p[(size_t)blockIdx.x * 256 + threadIdx.x] = make_uint4(0u, 0u, 0u, 0u);
}

// ---------------- demod: sumsq[n][f] += sum_c (w*COEF_W*s)^2 for one tap --------
__global__ __launch_bounds__(256) void demod_partial(
    const float* __restrict__ wgt, const float* __restrict__ s,
    float* __restrict__ sumsq)
{
    int f = blockIdx.x * 256 + threadIdx.x;   // grid.x = 2
    int n = blockIdx.y;                       // 8
    int t = blockIdx.z;                       // 9
    float acc = 0.f;
    for (int c = 0; c < 512; ++c) {
        float v = wgt[((size_t)t * 512 + c) * 512 + f] * (COEF_W * s[n * 512 + c]);
        acc += v * v;
    }
    atomicAdd(&sumsq[n * 512 + f], acc);
}

__global__ __launch_bounds__(256) void demod_finalize(float* __restrict__ d) {
    int i = blockIdx.x * 256 + threadIdx.x;   // 4096
    d[i] = rsqrtf(d[i] + 1e-8f);
}

// -------- wmod[n][t][f][c] = bf16(w[t][c][f] * COEF_W * s[n][c] * d[n][f]) ------
__global__ __launch_bounds__(256) void wmod_kernel(
    const float* __restrict__ wgt, const float* __restrict__ s,
    const float* __restrict__ d, unsigned short* __restrict__ wm)
{
    __shared__ float tile[64][65];
    int n = blockIdx.z, t = blockIdx.y;
    int ct = blockIdx.x & 7, ft = blockIdx.x >> 3;
    int cbase = ct * 64, fbase = ft * 64;
    int tid = threadIdx.x;
    int fi = tid & 63;
    for (int ci = tid >> 6; ci < 64; ci += 4) {
        int c = cbase + ci;
        tile[ci][fi] = wgt[((size_t)t * 512 + c) * 512 + fbase + fi] * (COEF_W * s[n * 512 + c]);
    }
    __syncthreads();
    int ci2 = tid & 63;
    for (int fi2 = tid >> 6; fi2 < 64; fi2 += 4) {
        int f = fbase + fi2;
        float v = tile[ci2][fi2] * d[n * 512 + f];
        wm[(((size_t)n * KT + t) * 512 + f) * 512 + cbase + ci2] = f2bf(v);
    }
}

// ------ xtp[n][h+1][w+1][c] = bf16(x[n][c][h][w])  (NCHW -> padded NHWC) --------
__global__ __launch_bounds__(256) void xt_kernel(
    const float* __restrict__ x, unsigned short* __restrict__ xtp)
{
    __shared__ float tile[64][65];
    int n = blockIdx.z;
    int pbase = blockIdx.x * 64;   // 64 pixel tiles
    int cbase = blockIdx.y * 64;   // 8 channel tiles
    int tid = threadIdx.x;
    int pi = tid & 63;
    for (int ci = tid >> 6; ci < 64; ci += 4)
        tile[ci][pi] = x[((size_t)n * 512 + cbase + ci) * HW + pbase + pi];
    __syncthreads();
    int ci2 = tid & 63;
    for (int pi2 = tid >> 6; pi2 < 64; pi2 += 4) {
        int p = pbase + pi2;
        int h = p >> 6, w = p & 63;
        xtp[(((size_t)n * HP + h + 1) * WP + w + 1) * CIN + cbase + ci2] =
            f2bf(tile[ci2][pi2]);
    }
}

// ---------------- main conv as implicit GEMM with bf16 WMMA ---------------------
// Workgroup: 256 thr = 8 waves (2 M x 4 N). Tile: 128 fmaps x 4 rows (256 pixels).
// Per wave: 64 fmaps x 64 pixels (one W-row) = 4x4 accumulator fragments.
// Zero-padded NHWC input -> no boundary predication in the K loop.
// K loop: 9 taps x 16 cin-chunks of 32; 16 WMMAs + 16 b128 loads per step,
// unrolled 4x for scheduling depth; next-tap streams prefetched L2->WGP$.
__global__ __launch_bounds__(256) void modconv_wmma(
    const unsigned short* __restrict__ xtp,  // [8][66][66][512] bf16, zero border
    const unsigned short* __restrict__ wm,   // [8][9][512][512] bf16
    float* __restrict__ out)                 // [8][512][4096] f32
{
    const int n    = blockIdx.z;
    const int lane = threadIdx.x & 31;
    const int wave = threadIdx.x >> 5;
    const int lrow = lane & 15;                 // row/col within a 16-wide fragment
    const int cinHalf = (lane >> 4) << 4;       // lanes 16-31 hold K=16..31 half

    const int fBase = blockIdx.x * 128 + (wave & 1) * 64;
    const int hRow  = blockIdx.y * 4 + (wave >> 1);   // output row owned by wave

    // Per-lane base pointers; everything else is wave-uniform or an imm offset.
    const unsigned short* wmLane =
        wm + (size_t)n * KT * FM * CIN + (size_t)(fBase + lrow) * CIN + cinHalf;
    const unsigned short* xtLane =
        xtp + (size_t)n * HP * WP * CIN + (size_t)lrow * CIN + cinHalf;

    v8f acc[4][4];
    #pragma unroll
    for (int i = 0; i < 4; ++i)
        #pragma unroll
        for (int j = 0; j < 4; ++j)
            acc[i][j] = (v8f)0.0f;

    for (int t = 0; t < KT; ++t) {
        const int ky = t / 3;                  // 0..2 (padded coords)
        const int kx = t % 3;
        const unsigned short* at = wmLane + t * (FM * CIN);
        const unsigned short* bt = xtLane + ((hRow + ky) * WP + kx) * CIN;

        // Warm next tap's streams (L2 -> WGP$); lowers to global_prefetch_b8.
        if (t + 1 < KT) {
            const int kyn = (t + 1) / 3, kxn = (t + 1) % 3;
            __builtin_prefetch(wmLane + (t + 1) * (FM * CIN), 0, 3);
            __builtin_prefetch(xtLane + ((hRow + kyn) * WP + kxn) * CIN, 0, 3);
        }

        #pragma unroll 4
        for (int kc = 0; kc < 16; ++kc) {
            const int co = kc * 32;
            v16bf a[4];
            #pragma unroll
            for (int fm = 0; fm < 4; ++fm)
                a[fm] = load_frag16(at + fm * (16 * CIN) + co);
            #pragma unroll
            for (int fn = 0; fn < 4; ++fn) {
                v16bf b = load_frag16(bt + fn * (16 * CIN) + co);
                #pragma unroll
                for (int fm = 0; fm < 4; ++fm)
                    acc[fm][fn] = __builtin_amdgcn_wmma_f32_16x16x32_bf16(
                        false, a[fm], false, b, (short)0, acc[fm][fn], false, false);
            }
        }
    }

    // D layout: VGPR v -> row v (lanes 0-15) / row v+8 (lanes 16-31), col = lane&15
    const int fRowOff = (lane >> 4) << 3;
    #pragma unroll
    for (int fm = 0; fm < 4; ++fm) {
        #pragma unroll
        for (int fn = 0; fn < 4; ++fn) {
            int pix = hRow * WW + fn * 16 + lrow;
            #pragma unroll
            for (int v = 0; v < 8; ++v) {
                int f = fBase + fm * 16 + fRowOff + v;
                out[((size_t)n * FM + f) * HW + pix] = acc[fm][fn][v];
            }
        }
    }
}

extern "C" void kernel_launch(void* const* d_in, const int* in_sizes, int n_in,
                              void* d_out, int out_size, void* d_ws, size_t ws_size,
                              hipStream_t stream) {
    (void)in_sizes; (void)n_in; (void)out_size; (void)ws_size;
    const float* x    = (const float*)d_in[0];   // [8,512,64,64]
    const float* dlat = (const float*)d_in[1];   // [8,512]
    const float* wgt  = (const float*)d_in[2];   // [3,3,512,512]
    const float* mw   = (const float*)d_in[3];   // [512,512]
    const float* mb   = (const float*)d_in[4];   // [512]
    float* out = (float*)d_out;                  // [8,512,64,64]

    // Workspace layout (total ~74 MB)
    float* s    = (float*)d_ws;                                  // 4096 f32
    float* dmod = s + 4096;                                      // 4096 f32
    unsigned short* wmod = (unsigned short*)(dmod + 4096);       // 8*9*512*512 bf16
    unsigned short* xtp  = wmod + (size_t)NB * KT * FM * CIN;    // 8*66*66*512 bf16

    const size_t xtpElems = (size_t)NB * HP * WP * CIN;          // 17,842,176
    const int    padBlocks = (int)(xtpElems * 2 / 16 / 256);     // 8712 (exact)

    style_kernel  <<<16, 256, 0, stream>>>(dlat, mw, mb, s);
    zero_f        <<<16, 256, 0, stream>>>(dmod);
    demod_partial <<<dim3(2, NB, KT), 256, 0, stream>>>(wgt, s, dmod);
    demod_finalize<<<16, 256, 0, stream>>>(dmod);
    wmod_kernel   <<<dim3(64, KT, NB), 256, 0, stream>>>(wgt, s, dmod, wmod);
    zero_u4       <<<padBlocks, 256, 0, stream>>>((uint4*)xtp);
    xt_kernel     <<<dim3(64, 8, NB), 256, 0, stream>>>(x, xtp);
    modconv_wmma  <<<dim3(FM / 128, HH / 4, NB), 256, 0, stream>>>(xtp, wmod, out);
}
